// FFJORD_38963943309672
// MI455X (gfx1250) — matrix operand order repaired and checked by reading, not compiled, not run on hardware
//
#include <hip/hip_runtime.h>
#include <math.h>

// ---------------------------------------------------------------------------
// Types for CDNA5 WMMA
// ---------------------------------------------------------------------------
typedef __attribute__((ext_vector_type(16))) _Float16 v16h;
typedef __attribute__((ext_vector_type(8)))  _Float16 v8h;
typedef __attribute__((ext_vector_type(8)))  float    v8f;

// LDS byte offset of a __shared__ object (generic -> AS3 addrspacecast)
__device__ __forceinline__ unsigned lds_offset(const void* p) {
    return (unsigned)(unsigned long long)
        (__attribute__((address_space(3))) const void*)p;
}

// CDNA5 async copy: global -> LDS, 16B per lane, tracked by ASYNCcnt
__device__ __forceinline__ void async_copy_b128(unsigned lds_byte_off,
                                                const void* gaddr) {
    asm volatile("global_load_async_to_lds_b128 %0, %1, off"
                 :
                 : "v"(lds_byte_off), "v"((unsigned long long)gaddr)
                 : "memory");
}
__device__ __forceinline__ void wait_async() {
    asm volatile("s_wait_asynccnt 0x0" ::: "memory");
}

// ---------------------------------------------------------------------------
// WMMA GEMM:  out = A[M x K] * Wt^T   (Wt stored [N][K], pre-transposed W)
// Block tile 128x128, 256 threads = 8 waves as 4(row) x 2(col);
// each wave: 32x64 = 8 accumulators -> 8 WMMAs per 32-wide K chunk.
// Global->LDS via async-to-LDS, double-buffered (2 x (8KB A + 8KB W)).
// K compile-time (576 / 512 / 64); all dims padded -> no bounds checks.
// Fused epilogues:
//   mode 0: f16 out = tanh(acc + bias)
//   mode 1: f32 out = acc + bias              (stage derivative k_x)
//   mode 2: f16 out = (1 - Hprev^2) * acc     (tangent through tanh)
//   mode 3: trace  : atomicAdd(tr[row], sum_col acc*eps)   (Hutchinson)
// ---------------------------------------------------------------------------
template <int KT>
__launch_bounds__(256)
__global__ void gemm_wmma_kernel(const _Float16* __restrict__ A, int lda,
                                 const _Float16* __restrict__ Wt,  // [N][KT]
                                 const float* __restrict__ bias, int biasN,
                                 const _Float16* __restrict__ Hprev,
                                 const float* __restrict__ eps, int epsD,
                                 void* __restrict__ outp, int ldo,
                                 float* __restrict__ traceOut,
                                 int mode)
{
    __shared__ __align__(16) _Float16 sA[2][128 * 32];
    __shared__ __align__(16) _Float16 sW[2][128 * 32];

    const int tid  = threadIdx.x;
    const int wave = tid >> 5;
    const int lane = tid & 31;
    const int wr   = wave >> 1;   // 0..3 : 32-row slice
    const int wc   = wave & 1;    // 0..1 : 64-col slice
    const int blockRow = blockIdx.x * 128;
    const int blockCol = blockIdx.y * 128;

    const unsigned sA_off = lds_offset(&sA[0][0]);
    const unsigned sW_off = lds_offset(&sW[0][0]);

    v8f acc[2][4] = {};           // [row frag][col frag]

    const int r16 = lane & 15;    // row/col within a 16-group
    const int kh  = lane >> 4;    // K-half owned by this lane

    // issue one 128x32 double tile (A + W) of async loads into buffer `buf`
    auto load_tile = [&](int buf, int k0) {
        #pragma unroll
        for (int p = 0; p < 2; ++p) {
            int idx = p * 256 + tid;
            int row = idx >> 2, seg = idx & 3;
            async_copy_b128(sA_off + (unsigned)((buf * 4096 + row * 32 + seg * 8) * 2),
                            &A[(size_t)(blockRow + row) * lda + k0 + seg * 8]);
        }
        #pragma unroll
        for (int p = 0; p < 2; ++p) {
            int idx = p * 256 + tid;
            int row = idx >> 2, seg = idx & 3;
            async_copy_b128(sW_off + (unsigned)((buf * 4096 + row * 32 + seg * 8) * 2),
                            &Wt[(size_t)(blockCol + row) * KT + k0 + seg * 8]);
        }
    };

    constexpr int NK = KT / 32;
    load_tile(0, 0);
    wait_async();
    __syncthreads();

    for (int kc = 0; kc < NK; ++kc) {
        const int cur = kc & 1;
        if (kc + 1 < NK) load_tile(cur ^ 1, (kc + 1) * 32);

        // A fragments (two 16x32 row tiles for this wave)
        v16h afrag[2];
        #pragma unroll
        for (int r = 0; r < 2; ++r) {
            const _Float16* pa =
                &sA[cur][(wr * 32 + r * 16 + r16) * 32 + kh * 8];
            v8h alo = *(const v8h*)(pa);
            v8h ahi = *(const v8h*)(pa + 16);
            #pragma unroll
            for (int i = 0; i < 8; ++i) { afrag[r][i] = alo[i]; afrag[r][i + 8] = ahi[i]; }
        }
        // 4 col fragments, 2 WMMAs each (reuse B across row frags)
        #pragma unroll
        for (int c = 0; c < 4; ++c) {
            const _Float16* pb =
                &sW[cur][(wc * 64 + c * 16 + r16) * 32 + kh * 8];
            v8h blo = *(const v8h*)(pb);
            v8h bhi = *(const v8h*)(pb + 16);
            v16h bfrag;
            #pragma unroll
            for (int i = 0; i < 8; ++i) { bfrag[i] = blo[i]; bfrag[i + 8] = bhi[i]; }
            #pragma unroll
            for (int r = 0; r < 2; ++r)
                acc[r][c] = __builtin_amdgcn_wmma_f32_16x16x32_f16(
                    false, afrag[r], false, bfrag, (short)0, acc[r][c], false, false);
        }

        if (kc + 1 < NK) wait_async();
        __syncthreads();
    }

    // ---- epilogue: D layout: VGPR v -> M = v + (lane>=16)*8, N = lane&15 ----
    const int lcol  = lane & 15;
    const int lhalf = lane >> 4;
    #pragma unroll
    for (int r = 0; r < 2; ++r) {
        #pragma unroll
        for (int c = 0; c < 4; ++c) {
            const int col = blockCol + wc * 64 + c * 16 + lcol;
            #pragma unroll
            for (int v = 0; v < 8; ++v) {
                const int row = blockRow + wr * 32 + r * 16 + lhalf * 8 + v;
                float val = acc[r][c][v];
                if (mode == 0) {
                    float b = (bias && col < biasN) ? bias[col] : 0.f;
                    ((_Float16*)outp)[(size_t)row * ldo + col] =
                        (_Float16)tanhf(val + b);
                } else if (mode == 1) {
                    float b = (bias && col < biasN) ? bias[col] : 0.f;
                    ((float*)outp)[(size_t)row * ldo + col] = val + b;
                } else if (mode == 2) {
                    float hp = (float)Hprev[(size_t)row * ldo + col];
                    ((_Float16*)outp)[(size_t)row * ldo + col] =
                        (_Float16)(val * (1.f - hp * hp));
                } else { // mode 3: Hutchinson trace partial row-dot
                    float e = (col < epsD) ? eps[(size_t)row * epsD + col] : 0.f;
                    float t = val * e;
                    t += __shfl_xor(t, 1, 16);
                    t += __shfl_xor(t, 2, 16);
                    t += __shfl_xor(t, 4, 16);
                    t += __shfl_xor(t, 8, 16);
                    if (lcol == 0) atomicAdd(&traceOut[row], t);
                }
            }
        }
    }
}

// ---------------------------------------------------------------------------
// Elementwise helper kernels
// ---------------------------------------------------------------------------
__global__ void fill_f32_kernel(float* __restrict__ p, float v, size_t n) {
    size_t i = (size_t)blockIdx.x * blockDim.x + threadIdx.x;
    if (i < n) p[i] = v;
}

__global__ void pack_x0_kernel(const float* __restrict__ x0, int D,
                               float* __restrict__ x, int Dp, size_t n) {
    size_t i = (size_t)blockIdx.x * blockDim.x + threadIdx.x;
    if (i >= n) return;
    int row = (int)(i / Dp), j = (int)(i % Dp);
    x[i] = (j < D) ? x0[(size_t)row * D + j] : 0.f;
}

__global__ void pack_eps_kernel(const float* __restrict__ eps, int D,
                                _Float16* __restrict__ E, int Kp, size_t n) {
    size_t i = (size_t)blockIdx.x * blockDim.x + threadIdx.x;
    if (i >= n) return;
    int row = (int)(i / Kp), j = (int)(i % Kp);
    E[i] = (j < D) ? (_Float16)eps[(size_t)row * D + j] : (_Float16)0.f;
}

__global__ void pack_in_kernel(const float* __restrict__ xi, int ldx, int D, float t,
                               const float* __restrict__ c1, int n1,
                               const float* __restrict__ c2, int n2,
                               _Float16* __restrict__ out, int Kp, size_t n) {
    size_t i = (size_t)blockIdx.x * blockDim.x + threadIdx.x;
    if (i >= n) return;
    int row = (int)(i / Kp), j = (int)(i % Kp);
    float v;
    if (j < D)                     v = xi[(size_t)row * ldx + j];
    else if (j == D)               v = t;
    else if (j < D + 1 + n1)       v = c1[(size_t)row * n1 + (j - D - 1)];
    else if (j < D + 1 + n1 + n2)  v = c2[(size_t)row * n2 + (j - D - 1 - n1)];
    else                           v = 0.f;
    out[i] = (_Float16)v;
}

__global__ void conv_wt_kernel(const float* __restrict__ W, int K, int N,
                               _Float16* __restrict__ Wt, int Kp, size_t n) {
    size_t i = (size_t)blockIdx.x * blockDim.x + threadIdx.x;
    if (i >= n) return;
    int nn = (int)(i / Kp), k = (int)(i % Kp);
    Wt[i] = (nn < N && k < K) ? (_Float16)W[(size_t)k * N + nn] : (_Float16)0.f;
}

__global__ void axpy5_kernel(const float* __restrict__ x, float* __restrict__ y, size_t n,
                             const float* k0, float c0, const float* k1, float c1,
                             const float* k2, float c2, const float* k3, float c3,
                             const float* k4, float c4) {
    size_t i = (size_t)blockIdx.x * blockDim.x + threadIdx.x;
    if (i >= n) return;
    float v = x[i];
    if (k0) v += c0 * k0[i];
    if (k1) v += c1 * k1[i];
    if (k2) v += c2 * k2[i];
    if (k3) v += c3 * k3[i];
    if (k4) v += c4 * k4[i];
    y[i] = v;
}

__global__ void lincomb6_kernel(float* __restrict__ dst, size_t n,
                                const float* __restrict__ k, size_t stride,
                                float c0, float c1, float c2,
                                float c3, float c4, float c5) {
    size_t i = (size_t)blockIdx.x * blockDim.x + threadIdx.x;
    if (i >= n) return;
    dst[i] += c0 * k[i] + c1 * k[stride + i] + c2 * k[2 * stride + i]
            + c3 * k[3 * stride + i] + c4 * k[4 * stride + i] + c5 * k[5 * stride + i];
}

__global__ void final_logp_kernel(const float* __restrict__ x, int Dp, int D,
                                  const float* __restrict__ L, int B,
                                  float* __restrict__ out) {
    int row = blockIdx.x * blockDim.x + threadIdx.x;
    if (row >= B) return;
    const float LOG2PI = 1.8378770664093453f;
    float s = 0.f;
    for (int j = 0; j < Dp; ++j) {
        float z = x[(size_t)row * Dp + j];
        s += z * z;
    }
    float lp = -0.5f * s - 0.5f * (float)D * LOG2PI + L[row];
    atomicAdd(out, -lp * (1.f / (float)B));
}

// ---------------------------------------------------------------------------
// Host orchestration
// ---------------------------------------------------------------------------
static inline size_t cdiv(size_t a, size_t b) { return (a + b - 1) / b; }

static void launch_gemm(hipStream_t s, int B, const _Float16* A, int lda,
                        const _Float16* Wt, int N, int K,
                        const float* bias, int biasN, const _Float16* Hp,
                        const float* eps, int epsD, void* outp, int ldo,
                        float* tr, int mode) {
    dim3 g(B / 128, N / 128);
    if (K == 576)
        gemm_wmma_kernel<576><<<g, 256, 0, s>>>(A, lda, Wt, bias, biasN, Hp, eps, epsD, outp, ldo, tr, mode);
    else if (K == 64)
        gemm_wmma_kernel<64><<<g, 256, 0, s>>>(A, lda, Wt, bias, biasN, Hp, eps, epsD, outp, ldo, tr, mode);
    else
        gemm_wmma_kernel<512><<<g, 256, 0, s>>>(A, lda, Wt, bias, biasN, Hp, eps, epsD, outp, ldo, tr, mode);
}

extern "C" void kernel_launch(void* const* d_in, const int* in_sizes, int n_in,
                              void* d_out, int out_size, void* d_ws, size_t ws_size,
                              hipStream_t stream) {
    (void)in_sizes; (void)n_in; (void)out_size; (void)ws_size;

    const float* voxel = (const float*)d_in[0];
    const float* energy = (const float*)d_in[1];
    const float* cond  = (const float*)d_in[2];
    const float* eps_v = (const float*)d_in[3];
    const float* eps_e = (const float*)d_in[4];
    const float* W1v = (const float*)d_in[5];  const float* b1v = (const float*)d_in[6];
    const float* W2v = (const float*)d_in[7];  const float* b2v = (const float*)d_in[8];
    const float* W3v = (const float*)d_in[9];  const float* b3v = (const float*)d_in[10];
    const float* W1e = (const float*)d_in[11]; const float* b1e = (const float*)d_in[12];
    const float* W2e = (const float*)d_in[13]; const float* b2e = (const float*)d_in[14];
    const float* W3e = (const float*)d_in[15]; const float* b3e = (const float*)d_in[16];

    const int B = 8192, D = 504, E = 45, H = 512;
    const int Dp = 512;            // padded voxel state width (mult of 128)
    const int K1v = D + 1 + E + 1; // 551
    const int K1vp = 576;
    const int Ep = 128;            // padded energy state width (mult of 128)
    const int K1e = E + 1 + 1;     // 47
    const int K1ep = 64;
    const int N_STEPS = 10;
    const float dt = -1.0f / (float)N_STEPS;

    // DOPRI5 tableau (first 6 stages, b7 = 0)
    const float C_[6] = {0.f, 0.2f, 0.3f, 0.8f, 8.f / 9.f, 1.f};
    const float A_[6][5] = {
        {0, 0, 0, 0, 0},
        {1.f / 5, 0, 0, 0, 0},
        {3.f / 40, 9.f / 40, 0, 0, 0},
        {44.f / 45, -56.f / 15, 32.f / 9, 0, 0},
        {19372.f / 6561, -25360.f / 2187, 64448.f / 6561, -212.f / 729, 0},
        {9017.f / 3168, -355.f / 33, 46732.f / 5247, 49.f / 176, -5103.f / 18656}};
    const float Bc[6] = {35.f / 384, 0.f, 500.f / 1113, 125.f / 192, -2187.f / 6784, 11.f / 84};

    // -------- workspace bump allocator --------
    char* wsb = (char*)d_ws;
    size_t off = 0;
    auto alloc = [&](size_t bytes) -> char* {
        char* p = wsb + off;
        off += (bytes + 255) & ~(size_t)255;
        return p;
    };
    _Float16* Wt1v = (_Float16*)alloc((size_t)H * K1vp * 2);
    _Float16* Wt2v = (_Float16*)alloc((size_t)H * H * 2);
    _Float16* Wt3v = (_Float16*)alloc((size_t)Dp * H * 2);
    _Float16* Wt1e = (_Float16*)alloc((size_t)H * K1ep * 2);
    _Float16* Wt2e = (_Float16*)alloc((size_t)H * H * 2);
    _Float16* Wt3e = (_Float16*)alloc((size_t)Ep * H * 2);
    _Float16* Ain  = (_Float16*)alloc((size_t)B * K1vp * 2);
    _Float16* Ein  = (_Float16*)alloc((size_t)B * K1vp * 2);
    _Float16* h1   = (_Float16*)alloc((size_t)B * H * 2);
    _Float16* h2   = (_Float16*)alloc((size_t)B * H * 2);
    _Float16* dh1  = (_Float16*)alloc((size_t)B * H * 2);
    _Float16* dh2  = (_Float16*)alloc((size_t)B * H * 2);
    float* xbuf = (float*)alloc((size_t)B * Dp * 4);
    float* xibuf = (float*)alloc((size_t)B * Dp * 4);
    float* kx = (float*)alloc((size_t)6 * B * Dp * 4);
    float* kl = (float*)alloc((size_t)6 * B * 4);
    float* Lb = (float*)alloc((size_t)B * 4);

    // -------- per-call weight conversion: f32 -> f16^T, zero padded ----------
    auto conv = [&](const float* W, int K, int N, _Float16* Wt, int Kp, int Np) {
        size_t n = (size_t)Np * Kp;
        conv_wt_kernel<<<cdiv(n, 256), 256, 0, stream>>>(W, K, N, Wt, Kp, n);
    };
    conv(W1v, K1v, H, Wt1v, K1vp, H);
    conv(W2v, H, H, Wt2v, H, H);
    conv(W3v, H, D, Wt3v, H, Dp);
    conv(W1e, K1e, H, Wt1e, K1ep, H);
    conv(W2e, H, H, Wt2e, H, H);
    conv(W3e, H, E, Wt3e, H, Ep);

    fill_f32_kernel<<<1, 1, 0, stream>>>((float*)d_out, 0.f, 1);

    // -------- one flow: integrate (x, L) from t=1 -> t=0, fixed DOPRI5 -------
    auto run_flow = [&](int Dd, int Ddp, int Kp, const float* x0,
                        const float* c1, int n1, const float* c2, int n2,
                        const float* eps,
                        const _Float16* Wt1, const float* b1,
                        const _Float16* Wt2, const float* b2,
                        const _Float16* Wt3, const float* b3) {
        const size_t nX = (size_t)B * Ddp;
        const size_t nA = (size_t)B * Kp;
        pack_x0_kernel<<<cdiv(nX, 256), 256, 0, stream>>>(x0, Dd, xbuf, Ddp, nX);
        fill_f32_kernel<<<cdiv((size_t)B, 256), 256, 0, stream>>>(Lb, 0.f, (size_t)B);
        pack_eps_kernel<<<cdiv(nA, 256), 256, 0, stream>>>(eps, Dd, Ein, Kp, nA);

        for (int nstep = 0; nstep < N_STEPS; ++nstep) {
            float t0 = 1.f + (float)nstep * dt;
            fill_f32_kernel<<<cdiv((size_t)6 * B, 256), 256, 0, stream>>>(kl, 0.f, (size_t)6 * B);

            for (int i = 0; i < 6; ++i) {
                const float* xcur = xbuf;
                if (i > 0) {
                    const float* kp[5] = {nullptr, nullptr, nullptr, nullptr, nullptr};
                    float cf[5] = {0, 0, 0, 0, 0};
                    for (int j = 0; j < i; ++j) {
                        kp[j] = kx + (size_t)j * B * Ddp;
                        cf[j] = dt * A_[i][j];
                    }
                    axpy5_kernel<<<cdiv(nX, 256), 256, 0, stream>>>(
                        xbuf, xibuf, nX, kp[0], cf[0], kp[1], cf[1], kp[2], cf[2],
                        kp[3], cf[3], kp[4], cf[4]);
                    xcur = xibuf;
                }
                float ti = t0 + C_[i] * dt;
                pack_in_kernel<<<cdiv(nA, 256), 256, 0, stream>>>(
                    xcur, Ddp, Dd, ti, c1, n1, c2, n2, Ain, Kp, nA);

                // primal pass
                launch_gemm(stream, B, Ain, Kp, Wt1, H, Kp, b1, H, nullptr,
                            nullptr, 0, h1, H, nullptr, 0);
                launch_gemm(stream, B, h1, H, Wt2, H, H, b2, H, nullptr,
                            nullptr, 0, h2, H, nullptr, 0);
                launch_gemm(stream, B, h2, H, Wt3, Ddp, H, b3, Dd, nullptr,
                            nullptr, 0, kx + (size_t)i * B * Ddp, Ddp, nullptr, 1);
                // tangent pass (JVP): d_in = [eps, 0...]
                launch_gemm(stream, B, Ein, Kp, Wt1, H, Kp, nullptr, 0, h1,
                            nullptr, 0, dh1, H, nullptr, 2);
                launch_gemm(stream, B, dh1, H, Wt2, H, H, nullptr, 0, h2,
                            nullptr, 0, dh2, H, nullptr, 2);
                launch_gemm(stream, B, dh2, H, Wt3, Ddp, H, nullptr, 0, nullptr,
                            eps, Dd, nullptr, Ddp, kl + (size_t)i * B, 3);
            }
            lincomb6_kernel<<<cdiv(nX, 256), 256, 0, stream>>>(
                xbuf, nX, kx, (size_t)B * Ddp,
                dt * Bc[0], dt * Bc[1], dt * Bc[2], dt * Bc[3], dt * Bc[4], dt * Bc[5]);
            lincomb6_kernel<<<cdiv((size_t)B, 256), 256, 0, stream>>>(
                Lb, (size_t)B, kl, (size_t)B,
                dt * Bc[0], dt * Bc[1], dt * Bc[2], dt * Bc[3], dt * Bc[4], dt * Bc[5]);
        }
        final_logp_kernel<<<cdiv((size_t)B, 256), 256, 0, stream>>>(
            xbuf, Ddp, Dd, Lb, B, (float*)d_out);
    };

    // voxel flow: cond = [energy, cond]
    run_flow(D, Dp, K1vp, voxel, energy, E, cond, 1, eps_v,
             Wt1v, b1v, Wt2v, b2v, Wt3v, b3v);
    // energy flow: cond = [cond]
    run_flow(E, Ep, K1ep, energy, cond, 1, nullptr, 0, eps_e,
             Wt1e, b1e, Wt2e, b2e, Wt3e, b3e);
}
